// LinearDeepSeekV3_5282809775018
// MI455X (gfx1250) — 8-bit matrix operand order repaired and checked by reading, compiled, not run on hardware
//
#include <hip/hip_runtime.h>
#include <hip/hip_bf16.h>
#include <math.h>

#define MDIM 4096
#define KDIM 7168
#define NDIM 4096
#define KB   (KDIM / 128)
#define FP8_MAX 448.0f

typedef __attribute__((ext_vector_type(16))) int      v16i;
typedef __attribute__((ext_vector_type(8)))  float    v8f;
typedef __attribute__((ext_vector_type(4)))  unsigned u32x4;
typedef __attribute__((ext_vector_type(4)))  int      i32x4;
typedef __attribute__((ext_vector_type(8)))  int      i32x8;

#if __has_builtin(__builtin_amdgcn_tensor_load_to_lds) && __has_builtin(__builtin_amdgcn_s_wait_tensorcnt)
#define USE_TDM 1
#else
#define USE_TDM 0
#endif

#if __has_builtin(__builtin_amdgcn_cvt_pk_fp8_f32)
__device__ __forceinline__ unsigned f32x4_to_fp8x4(float a, float b, float c, float d) {
  int lo  = __builtin_amdgcn_cvt_pk_fp8_f32(a, b, 0, false);
  int all = __builtin_amdgcn_cvt_pk_fp8_f32(c, d, lo, true);
  return (unsigned)all;
}
#else
__device__ __forceinline__ unsigned sw_e4m3(float f) {
  float a = fminf(fabsf(f), FP8_MAX);
  unsigned s = (f < 0.0f) ? 0x80u : 0x00u;
  if (a < 9.765625e-4f) return s;
  int e; float m = frexpf(a, &e);
  int exp = e - 1 + 7;
  int mant;
  if (exp <= 0) { mant = (int)(ldexpf(a, 9) + 0.5f); exp = 0; if (mant > 7) { mant = 0; exp = 1; } }
  else { mant = (int)((m * 2.0f - 1.0f) * 8.0f + 0.5f); if (mant > 7) { mant = 0; exp++; } }
  if (exp > 15) { exp = 15; mant = 6; }
  return s | (unsigned)(exp << 3) | (unsigned)mant;
}
__device__ __forceinline__ unsigned f32x4_to_fp8x4(float a, float b, float c, float d) {
  return sw_e4m3(a) | (sw_e4m3(b) << 8) | (sw_e4m3(c) << 16) | (sw_e4m3(d) << 24);
}
#endif

__global__ __launch_bounds__(256)
void act_quant_kernel(const float* __restrict__ x,
                      unsigned char* __restrict__ x8,
                      float* __restrict__ ascale_t) {
  const int  lane = threadIdx.x & 31;
  const long long wid = ((long long)blockIdx.x * blockDim.x + threadIdx.x) >> 5;
  const long long total = (long long)MDIM * KB;
  if (wid >= total) return;
  const int m  = (int)(wid / KB);
  const int kb = (int)(wid % KB);

  const float4 v = *(const float4*)(x + (size_t)m * KDIM + kb * 128 + lane * 4);
  float amax = fmaxf(fmaxf(fabsf(v.x), fabsf(v.y)), fmaxf(fabsf(v.z), fabsf(v.w)));
  #pragma unroll
  for (int off = 16; off > 0; off >>= 1)
    amax = fmaxf(amax, __shfl_xor(amax, off, 32));

  const float scale = fmaxf(amax, 1e-12f) * (1.0f / FP8_MAX);
  const float r = 1.0f / scale;
  const unsigned packed = f32x4_to_fp8x4(v.x * r, v.y * r, v.z * r, v.w * r);
  *(unsigned*)(x8 + (size_t)m * KDIM + kb * 128 + lane * 4) = packed;
  if (lane == 0) ascale_t[(size_t)kb * MDIM + m] = scale;
}

__global__ __launch_bounds__(256)
void weight_pack_kernel(const float* __restrict__ w, unsigned char* __restrict__ w8) {
  const size_t i = ((size_t)blockIdx.x * blockDim.x + threadIdx.x) * 4;
  if (i >= (size_t)NDIM * KDIM) return;
  const float4 v = *(const float4*)(w + i);
  *(unsigned*)(w8 + i) = f32x4_to_fp8x4(v.x, v.y, v.z, v.w);
}

#define BM 128
#define BN 64
#define LDS_STRIDE 144
#define A_BYTES (BM * LDS_STRIDE)
#define B_BYTES (BN * LDS_STRIDE)
#define BUF_BYTES (A_BYTES + B_BYTES)

#if USE_TDM
__device__ __forceinline__ void tdm_load_2d(unsigned lds_off,
                                            const unsigned char* gaddr,
                                            unsigned rows, unsigned tensor_rows) {
  const unsigned long long ga = (unsigned long long)(size_t)gaddr;
  u32x4 g0;
  g0[0] = 1u;
  g0[1] = lds_off;
  g0[2] = (unsigned)ga;
  g0[3] = (unsigned)((ga >> 32) & 0x01FFFFFFu) | (2u << 30);
  i32x8 g1;
  g1[0] = (int)((1u << 20) | (4u << 22) | (3u << 25));
  g1[1] = (int)(((unsigned)KDIM & 0xFFFFu) << 16);
  g1[2] = (int)((((unsigned)KDIM >> 16) & 0xFFFFu) | ((tensor_rows & 0xFFFFu) << 16));
  g1[3] = (int)(((tensor_rows >> 16) & 0xFFFFu) | (128u << 16));
  g1[4] = (int)(rows & 0xFFFFu);
  g1[5] = (int)KDIM;
  g1[6] = 0;
  g1[7] = 0;
  i32x4 gz = {0, 0, 0, 0};
#if defined(__clang_major__) && (__clang_major__ >= 23)
  i32x8 gz8 = {0, 0, 0, 0, 0, 0, 0, 0};
  __builtin_amdgcn_tensor_load_to_lds(g0, g1, gz, gz, gz8, 0);
#else
  __builtin_amdgcn_tensor_load_to_lds(g0, g1, gz, gz, 0);
#endif
}
#endif

__device__ __forceinline__ void stage_tiles_manual(unsigned char* lbuf,
                                                   const unsigned char* __restrict__ x8,
                                                   const unsigned char* __restrict__ w8,
                                                   int m_base, int n_base, int kb, int tid) {
  #pragma unroll
  for (int it = 0; it < 4; ++it) {
    const int idx = it * 256 + tid;
    const int row = idx >> 3, ch = idx & 7;
    const uint4 d = *(const uint4*)(x8 + (size_t)(m_base + row) * KDIM + kb * 128 + ch * 16);
    *(uint4*)(lbuf + row * LDS_STRIDE + ch * 16) = d;
  }
  #pragma unroll
  for (int it = 0; it < 2; ++it) {
    const int idx = it * 256 + tid;
    const int row = idx >> 3, ch = idx & 7;
    const uint4 d = *(const uint4*)(w8 + (size_t)(n_base + row) * KDIM + kb * 128 + ch * 16);
    *(uint4*)(lbuf + A_BYTES + row * LDS_STRIDE + ch * 16) = d;
  }
}

__global__ __launch_bounds__(256)
void gemm_fp8_blockscale_kernel(const unsigned char* __restrict__ x8,
                                const unsigned char* __restrict__ w8,
                                const float* __restrict__ asct,
                                const float* __restrict__ wscale,
                                float* __restrict__ out) {
  __shared__ __align__(16) unsigned char lds[2 * BUF_BYTES];

  const int tid  = threadIdx.x;
  const int lane = tid & 31;
  const int wave = tid >> 5;
  const int wm   = (wave & 3) * 32;
  const int wn   = (wave >> 2) * 32;
  const int half = lane >> 4;
  const int l16  = lane & 15;

  const int m_base = blockIdx.x * BM;
  const int n_base = blockIdx.y * BN;
  const float* wsc_row = wscale + (size_t)(n_base >> 7) * KB;

  float acc[2][2][8];
  #pragma unroll
  for (int i = 0; i < 2; ++i)
    #pragma unroll
    for (int j = 0; j < 2; ++j)
      #pragma unroll
      for (int r = 0; r < 8; ++r) acc[i][j][r] = 0.0f;

#if USE_TDM
  const unsigned lds_base = (unsigned)(size_t)&lds[0];
  if (wave == 0) {
    tdm_load_2d(lds_base,            x8 + (size_t)m_base * KDIM, BM, MDIM);
    tdm_load_2d(lds_base + A_BYTES,  w8 + (size_t)n_base * KDIM, BN, NDIM);
  }
#else
  stage_tiles_manual(lds, x8, w8, m_base, n_base, 0, tid);
#endif

  for (int kb = 0; kb < KB; ++kb) {
    const int cur = kb & 1;
    unsigned char* lbuf = lds + cur * BUF_BYTES;

#if USE_TDM
    if (wave == 0) __builtin_amdgcn_s_wait_tensorcnt(0);
#endif
    __syncthreads();

    v16i afr[2], bfr[2];
    #pragma unroll
    for (int i = 0; i < 2; ++i) {
      const unsigned char* base = lbuf + (wm + i * 16 + l16) * LDS_STRIDE;
      #pragma unroll
      for (int vv = 0; vv < 16; ++vv) {
        const int c = vv >> 3, q = vv & 7;
        const int k = (vv >> 2) * 32 + (vv & 3) * 4 + half * 16;
        afr[i][vv] = *(const int*)(base + k);
      }
    }
    #pragma unroll
    for (int j = 0; j < 2; ++j) {
      const unsigned char* base = lbuf + A_BYTES + (wn + j * 16 + l16) * LDS_STRIDE + half * 16;
      #pragma unroll
      for (int g = 0; g < 4; ++g) {
        const uint4 d = *(const uint4*)(base + g * 32);
        bfr[j][g * 4 + 0] = (int)d.x;
        bfr[j][g * 4 + 1] = (int)d.y;
        bfr[j][g * 4 + 2] = (int)d.z;
        bfr[j][g * 4 + 3] = (int)d.w;
      }
    }
    __syncthreads();

    if (kb + 1 < KB) {
#if USE_TDM
      if (wave == 0) {
        const unsigned boff = lds_base + (1 - cur) * BUF_BYTES;
        tdm_load_2d(boff,           x8 + (size_t)m_base * KDIM + (size_t)(kb + 1) * 128, BM, MDIM);
        tdm_load_2d(boff + A_BYTES, w8 + (size_t)n_base * KDIM + (size_t)(kb + 1) * 128, BN, NDIM);
      }
#else
      stage_tiles_manual(lds + (1 - cur) * BUF_BYTES, x8, w8, m_base, n_base, kb + 1, tid);
#endif
      __builtin_prefetch(asct + (size_t)(kb + 1) * MDIM + m_base + (tid & 127), 0, 0);
    }

    const float w_s = wsc_row[kb];
    float asc[2][8];
    #pragma unroll
    for (int i = 0; i < 2; ++i) {
      const float* ap = asct + (size_t)kb * MDIM + (m_base + wm + i * 16 + half * 8);
      const float4 s0 = *(const float4*)ap;
      const float4 s1 = *(const float4*)(ap + 4);
      asc[i][0] = s0.x * w_s; asc[i][1] = s0.y * w_s;
      asc[i][2] = s0.z * w_s; asc[i][3] = s0.w * w_s;
      asc[i][4] = s1.x * w_s; asc[i][5] = s1.y * w_s;
      asc[i][6] = s1.z * w_s; asc[i][7] = s1.w * w_s;
    }

    #pragma unroll
    for (int i = 0; i < 2; ++i) {
      #pragma unroll
      for (int j = 0; j < 2; ++j) {
        v8f czero = {};
        v8f d = __builtin_amdgcn_wmma_f32_16x16x128_fp8_fp8(
            afr[i], bfr[j], (short)0, czero, false, false);
        #pragma unroll
        for (int r = 0; r < 8; ++r)
          acc[i][j][r] = fmaf(d[r], asc[i][r], acc[i][j][r]);
      }
    }
  }

  #pragma unroll
  for (int i = 0; i < 2; ++i) {
    #pragma unroll
    for (int j = 0; j < 2; ++j) {
      const int col = n_base + wn + j * 16 + l16;
      #pragma unroll
      for (int r = 0; r < 8; ++r) {
        const int row = m_base + wm + i * 16 + half * 8 + r;
        out[(size_t)row * NDIM + col] = acc[i][j][r];
      }
    }
  }
}

extern "C" void kernel_launch(void* const* d_in, const int* in_sizes, int n_in,
                              void* d_out, int out_size, void* d_ws, size_t ws_size,
                              hipStream_t stream) {
  const float* x   = (const float*)d_in[0];
  const float* w   = (const float*)d_in[1];
  const float* wsc = (const float*)d_in[2];
  float* out = (float*)d_out;

  unsigned char* x8 = (unsigned char*)d_ws;
  unsigned char* w8 = x8 + (size_t)MDIM * KDIM;
  float* asct = (float*)(w8 + (size_t)NDIM * KDIM);

  {
    const long long threads = (long long)MDIM * KB * 32;
    const int block = 256;
    const long long grid = (threads + block - 1) / block;
    hipLaunchKernelGGL(act_quant_kernel, dim3((unsigned)grid), dim3(block), 0, stream,
                       x, x8, asct);
  }
  {
    const size_t quads = (size_t)NDIM * KDIM / 4;
    const int block = 256;
    const size_t grid = (quads + block - 1) / block;
    hipLaunchKernelGGL(weight_pack_kernel, dim3((unsigned)grid), dim3(block), 0, stream,
                       w, w8);
  }
  {
    dim3 grid(MDIM / BM, NDIM / BN);
    hipLaunchKernelGGL(gemm_fp8_blockscale_kernel, grid, dim3(256), 0, stream,
                       x8, w8, asct, wsc, out);
  }
}
